// LlamaAttention_78640851189982
// MI455X (gfx1250) — matrix operand order repaired and checked by reading, not compiled, not run on hardware
//
#include <hip/hip_runtime.h>

typedef __bf16 bf16_t;
typedef bf16_t v16bf __attribute__((ext_vector_type(16)));
typedef bf16_t v8bf  __attribute__((ext_vector_type(8)));
typedef float  v8f   __attribute__((ext_vector_type(8)));
typedef unsigned short ushort_t;

#define BM 128
#define BN 128
#define BK 32
#define LDSS 40   // padded row stride (bf16 elements): 32 + 8 pad -> 80B rows, 16B aligned

__device__ __forceinline__ ushort_t f32_to_bf16(float f) {
    union { float f; unsigned u; } x; x.f = f;
    unsigned u = x.u;
    u += 0x7fffu + ((u >> 16) & 1u);       // round to nearest even
    return (ushort_t)(u >> 16);
}

__device__ __forceinline__ v8f zero8() {
    v8f z;
    #pragma unroll
    for (int i = 0; i < 8; ++i) z[i] = 0.0f;
    return z;
}

// 32-bit LDS byte offset of a __shared__ object (generic -> AS3 strips aperture)
__device__ __forceinline__ unsigned lds_off(const void* p) {
    return (unsigned)(unsigned long long)
        (const __attribute__((address_space(3))) void*)p;
}

// async DMA: 16 bytes per lane, global -> LDS, tracked by ASYNCcnt
__device__ __forceinline__ void async_copy_b128(unsigned lds_byte_off,
                                                const void* gptr) {
    asm volatile("global_load_async_to_lds_b128 %0, %1, off"
                 :: "v"(lds_byte_off), "v"((unsigned long long)gptr)
                 : "memory");
}

__device__ __forceinline__ void wait_asynccnt0() {
    asm volatile("s_wait_asynccnt 0x0" ::: "memory");
}

// ---------------------------------------------------------------------------
// fp32 -> bf16 conversion
// ---------------------------------------------------------------------------
__global__ void cvt_f32_bf16(const float* __restrict__ in,
                             ushort_t* __restrict__ out, long long n) {
    long long i = (long long)blockIdx.x * blockDim.x + threadIdx.x;
    long long stride = (long long)gridDim.x * blockDim.x;
    for (; i < n; i += stride) out[i] = f32_to_bf16(in[i]);
}

// ---------------------------------------------------------------------------
// NT GEMM: C[M,N] = alpha * A[M,K] * B[N,K]^T (+ mask[M,N])
// OUT_MODE: 0 = bf16, 1 = f32, 2 = bf16 stored transposed (C[N-major])
// Double-buffered LDS fed by async global->LDS DMA (ASYNCcnt).
// ---------------------------------------------------------------------------
template<int OUT_MODE, bool USE_MASK>
__global__ __launch_bounds__(256)
void gemm_nt(const ushort_t* __restrict__ A, const ushort_t* __restrict__ B,
             void* __restrict__ Cout, const float* __restrict__ mask,
             int M, int N, int K, int lda, int ldb, int ldc,
             long long sA, long long sB, long long sC, long long sMask,
             float alpha)
{
    __shared__ __align__(16) ushort_t sAt[2][BM * LDSS];
    __shared__ __align__(16) ushort_t sBt[2][BN * LDSS];

    const int b = blockIdx.z;
    A += (long long)b * sA;
    B += (long long)b * sB;

    const int tile_n = blockIdx.x * BN;
    const int tile_m = blockIdx.y * BM;

    const int tid  = threadIdx.x;
    const int wave = tid >> 5;
    const int lane = tid & 31;
    const int wm   = wave >> 2;   // 0..1 : 64-row slab
    const int wn   = wave & 3;    // 0..3 : 32-col slab

    v8f acc[4][2];
    #pragma unroll
    for (int i = 0; i < 4; ++i)
        #pragma unroll
        for (int j = 0; j < 2; ++j) acc[i][j] = zero8();

    // cooperative loaders: 4 threads/row * 8 bf16 (16B) each, two 64-row passes
    const int lr = tid >> 2;          // 0..63
    const int lc = (tid & 3) * 8;     // 0,8,16,24

    const int lane_r  = lane & 15;
    const int lane_kh = (lane >> 4) * 8;    // A frag K-half offset: 0 or 8
    const int lane_kb = (lane >> 4) * 16;   // unused; B frags also use lane_kh

    const ushort_t* Ap0 = &A[(long long)(tile_m + lr) * lda + lc];
    const ushort_t* Ap1 = &A[(long long)(tile_m + lr + 64) * lda + lc];
    const ushort_t* Bp0 = &B[(long long)(tile_n + lr) * ldb + lc];
    const ushort_t* Bp1 = &B[(long long)(tile_n + lr + 64) * ldb + lc];

    const unsigned la0 = lds_off(&sAt[0][lr * LDSS + lc]);
    const unsigned la1 = lds_off(&sAt[0][(lr + 64) * LDSS + lc]);
    const unsigned lb0 = lds_off(&sBt[0][lr * LDSS + lc]);
    const unsigned lb1 = lds_off(&sBt[0][(lr + 64) * LDSS + lc]);
    const unsigned bufstep = (unsigned)(BM * LDSS * sizeof(ushort_t));

    auto async_tile = [&](int k0, int buf) {
        const unsigned d = buf ? bufstep : 0u;
        async_copy_b128(la0 + d, Ap0 + k0);
        async_copy_b128(la1 + d, Ap1 + k0);
        async_copy_b128(lb0 + d, Bp0 + k0);
        async_copy_b128(lb1 + d, Bp1 + k0);
    };

    // prologue: DMA tile 0 into buffer 0
    async_tile(0, 0);
    wait_asynccnt0();
    __syncthreads();

    const int niter = K / BK;
    for (int it = 0; it < niter; ++it) {
        const int  cur  = it & 1;
        const bool more = (it + 1) < niter;

        // DMA next K-tile into the other buffer; safe: the barrier ending the
        // previous iteration guarantees all waves finished reading it.
        if (more) {
            async_tile((it + 1) * BK, cur ^ 1);
            if (it + 2 < niter) {   // global_prefetch_b8 of tile after next
                __builtin_prefetch(Ap0 + (it + 2) * BK, 0, 0);
                __builtin_prefetch(Bp0 + (it + 2) * BK, 0, 0);
            }
        }

        // A fragments (16x32): lane<16 holds K{kh..kh+7, 16+kh..}, per ISA layout
        v16bf afrag[4];
        #pragma unroll
        for (int i = 0; i < 4; ++i) {
            const ushort_t* p = &sAt[cur][(wm * 64 + i * 16 + lane_r) * LDSS];
            v16bf f;
            *(v8bf*)&f       = *(const v8bf*)&p[lane_kh];
            *((v8bf*)&f + 1) = *(const v8bf*)&p[lane_kh + 16];
            afrag[i] = f;
        }
        // B fragments (32x16): lane half holds K{kh..kh+7, 16+kh..}, same as A
        v16bf bfrag[2];
        #pragma unroll
        for (int j = 0; j < 2; ++j) {
            const ushort_t* p = &sBt[cur][(wn * 32 + j * 16 + lane_r) * LDSS];
            v16bf f;
            *(v8bf*)&f       = *(const v8bf*)&p[lane_kh];
            *((v8bf*)&f + 1) = *(const v8bf*)&p[lane_kh + 16];
            bfrag[j] = f;
        }

        #pragma unroll
        for (int i = 0; i < 4; ++i)
            #pragma unroll
            for (int j = 0; j < 2; ++j)
                acc[i][j] = __builtin_amdgcn_wmma_f32_16x16x32_bf16(
                    false, afrag[i], false, bfrag[j],
                    (short)0, acc[i][j], false, false);

        if (more) {
            wait_asynccnt0();   // own DMA writes to buffer cur^1 have landed
            __syncthreads();    // everyone's reads of cur + DMA writes done
        }
    }

    // epilogue: C layout -> VGPR r: M = (lane>=16 ? 8:0)+r, N = lane&15
    const int cn  = lane & 15;
    const int cm0 = (lane >> 4) * 8;
    #pragma unroll
    for (int i = 0; i < 4; ++i) {
        #pragma unroll
        for (int j = 0; j < 2; ++j) {
            const int gm_base = tile_m + wm * 64 + i * 16;
            const int gn      = tile_n + wn * 32 + j * 16 + cn;
            #pragma unroll
            for (int r = 0; r < 8; ++r) {
                const int gm = gm_base + cm0 + r;
                float v = acc[i][j][r] * alpha;
                if (USE_MASK)
                    v += mask[(long long)b * sMask + (long long)gm * N + gn];
                if (OUT_MODE == 1) {
                    ((float*)Cout)[(long long)b * sC + (long long)gm * ldc + gn] = v;
                } else if (OUT_MODE == 0) {
                    ((ushort_t*)Cout)[(long long)b * sC + (long long)gm * ldc + gn] =
                        f32_to_bf16(v);
                } else { // transposed store (used for V -> V^T)
                    ((ushort_t*)Cout)[(long long)b * sC + (long long)gn * ldc + gm] =
                        f32_to_bf16(v);
                }
            }
        }
    }
}

// ---------------------------------------------------------------------------
// row softmax: fp32 scores -> bf16 probs, one 256-thread block per row
// ---------------------------------------------------------------------------
__global__ __launch_bounds__(256)
void softmax_rows(const float* __restrict__ S, ushort_t* __restrict__ P, int ncols) {
    __shared__ float red[256];
    const long long base = (long long)blockIdx.x * ncols;
    const int tid = threadIdx.x;

    float m = -3.402823e38f;
    for (int c = tid; c < ncols; c += 256) m = fmaxf(m, S[base + c]);
    red[tid] = m; __syncthreads();
    for (int s = 128; s > 0; s >>= 1) {
        if (tid < s) red[tid] = fmaxf(red[tid], red[tid + s]);
        __syncthreads();
    }
    const float rowmax = red[0]; __syncthreads();

    float sum = 0.0f;
    for (int c = tid; c < ncols; c += 256) sum += __expf(S[base + c] - rowmax);
    red[tid] = sum; __syncthreads();
    for (int s = 128; s > 0; s >>= 1) {
        if (tid < s) red[tid] += red[tid + s];
        __syncthreads();
    }
    const float inv = 1.0f / red[0];

    for (int c = tid; c < ncols; c += 256)
        P[base + c] = f32_to_bf16(__expf(S[base + c] - rowmax) * inv);
}

// ---------------------------------------------------------------------------
// host orchestration
// ---------------------------------------------------------------------------
extern "C" void kernel_launch(void* const* d_in, const int* in_sizes, int n_in,
                              void* d_out, int out_size, void* d_ws, size_t ws_size,
                              hipStream_t stream) {
    (void)in_sizes; (void)n_in; (void)out_size; (void)ws_size;

    constexpr int B = 2, S = 4096, H = 2048;
    constexpr long long BS = (long long)B * S;          // 8192
    const float* X    = (const float*)d_in[0];
    const float* mask = (const float*)d_in[1];
    const float* Wq   = (const float*)d_in[2];
    const float* Wk   = (const float*)d_in[3];
    const float* Wv   = (const float*)d_in[4];
    const float* Wo   = (const float*)d_in[5];
    float* out = (float*)d_out;

    char* ws = (char*)d_ws;
    // byte offsets into workspace
    ushort_t* Xb  = (ushort_t*)(ws + 0);                       // [8192,2048] bf16
    ushort_t* Wqb = (ushort_t*)(ws + 33554432LL);
    ushort_t* Wkb = (ushort_t*)(ws + 41943040LL);
    ushort_t* Wvb = (ushort_t*)(ws + 50331648LL);
    ushort_t* Wob = (ushort_t*)(ws + 58720256LL);
    ushort_t* Qb  = (ushort_t*)(ws + 67108864LL);              // [8192,2048]
    ushort_t* Kb  = (ushort_t*)(ws + 100663296LL);             // [8192,2048]
    ushort_t* Vt  = (ushort_t*)(ws + 134217728LL);             // [2048,8192] (V^T)
    float*    Sc  = (float*)   (ws + 167772160LL);             // [2,4096,4096] f32
    ushort_t* Pb  = (ushort_t*)(ws + 301989888LL);             // [2,4096,4096]
    ushort_t* Cb  = (ushort_t*)(ws + 369098752LL);             // [2,4096,2048]

    const dim3 blk(256);

    // 1) fp32 -> bf16
    cvt_f32_bf16<<<4096, blk, 0, stream>>>(X,  Xb,  BS * H);
    cvt_f32_bf16<<<1024, blk, 0, stream>>>(Wq, Wqb, (long long)H * H);
    cvt_f32_bf16<<<1024, blk, 0, stream>>>(Wk, Wkb, (long long)H * H);
    cvt_f32_bf16<<<1024, blk, 0, stream>>>(Wv, Wvb, (long long)H * H);
    cvt_f32_bf16<<<1024, blk, 0, stream>>>(Wo, Wob, (long long)H * H);

    // 2) projections: Q,K = Xb * W^T ; V stored transposed -> Vt[2048,8192]
    {
        dim3 grid(H / BN, (int)(BS / BM), 1);
        gemm_nt<0, false><<<grid, blk, 0, stream>>>(Xb, Wqb, Qb, nullptr,
            (int)BS, H, H, H, H, H, 0, 0, 0, 0, 1.0f);
        gemm_nt<0, false><<<grid, blk, 0, stream>>>(Xb, Wkb, Kb, nullptr,
            (int)BS, H, H, H, H, H, 0, 0, 0, 0, 1.0f);
        gemm_nt<2, false><<<grid, blk, 0, stream>>>(Xb, Wvb, Vt, nullptr,
            (int)BS, H, H, H, H, (int)BS /*ldc of V^T*/, 0, 0, 0, 0, 1.0f);
    }

    // 3) scores = (Q K^T) * 1/sqrt(128) + mask   (fp32)
    {
        dim3 grid(S / BN, S / BM, B);
        gemm_nt<1, true><<<grid, blk, 0, stream>>>(Qb, Kb, Sc, mask,
            S, S, H, H, H, S,
            (long long)S * H, (long long)S * H,
            (long long)S * S, (long long)S * S,
            0.08838834764831845f);
    }

    // 4) softmax rows -> bf16 probs
    softmax_rows<<<B * S, blk, 0, stream>>>(Sc, Pb, S);

    // 5) ctx = P * V  (B operand is V^T rows = hidden dim; batch offset = b*S cols)
    {
        dim3 grid(H / BN, S / BM, B);
        gemm_nt<0, false><<<grid, blk, 0, stream>>>(Pb, Vt, Cb, nullptr,
            S, H, S, S, (int)BS /*ldb*/, H,
            (long long)S * S, (long long)S /*col offset per batch*/,
            (long long)S * H, 0, 1.0f);
    }

    // 6) out = ctx * Wo^T  (fp32 straight to d_out)
    {
        dim3 grid(H / BN, (int)(BS / BM), 1);
        gemm_nt<1, false><<<grid, blk, 0, stream>>>(Cb, Wob, out, nullptr,
            (int)BS, H, H, H, H, H, 0, 0, 0, 0, 1.0f);
    }
}
